// Attention_55121610277179
// MI455X (gfx1250) — matrix operand order repaired and checked by reading, not compiled, not run on hardware
//
#include <hip/hip_runtime.h>
#include <hip/hip_bf16.h>

// ---------------------------------------------------------------------------
// Types for CDNA5 WMMA (wave32) and TDM descriptors.
// ---------------------------------------------------------------------------
typedef __attribute__((ext_vector_type(16))) __bf16 v16bf;
typedef __attribute__((ext_vector_type(8)))  __bf16 v8bf;
typedef __attribute__((ext_vector_type(8)))  float  v8f;
typedef __attribute__((ext_vector_type(4)))  unsigned int u32x4;
typedef __attribute__((ext_vector_type(8)))  int i32x8;
typedef __attribute__((ext_vector_type(4)))  int i32x4;

#define B_DIM 8
#define T_DIM 2048
#define E_DIM 1024
#define S_DIM 512
#define M_TOT (B_DIM * T_DIM)   // 16384

__device__ __forceinline__ unsigned short f32_to_bf16(float f) {
  unsigned int u = __builtin_bit_cast(unsigned int, f);
  unsigned int r = u + 0x7FFFu + ((u >> 16) & 1u);   // round-to-nearest-even
  return (unsigned short)(r >> 16);
}

// ---------------------------------------------------------------------------
// TDM: 2D tile load Global -> LDS. data_size = 8B units. tile_dim0 = one row
// of `tile_d0_u8` 8B-units, `rows` rows, row stride `stride_u8` 8B-units.
// Descriptor bit layout per CDNA5 ISA (groups 0/1).
// ---------------------------------------------------------------------------
__device__ __forceinline__ void tdm_load_2d(unsigned lds_off, const void* gsrc,
                                            unsigned tile_d0_u8, unsigned rows,
                                            unsigned stride_u8) {
  union { unsigned w[4]; u32x4 v; } g0 = {};
  union { unsigned w[8]; i32x8 v; } g1 = {};
  unsigned long long ga = (unsigned long long)(size_t)gsrc;
  g0.w[0] = 1u;                                              // count = 1 (valid)
  g0.w[1] = lds_off;                                         // lds_addr
  g0.w[2] = (unsigned)ga;                                    // global_addr lo
  g0.w[3] = (unsigned)((ga >> 32) & 0x1FFFFFFu) | (2u << 30); // ga hi | type=2
  g1.w[0] = 3u << 16;                                        // data_size = 8B
  g1.w[1] = (stride_u8 & 0xFFFFu) << 16;                     // tensor_dim0 lo16
  g1.w[2] = (stride_u8 >> 16) | ((rows & 0xFFFFu) << 16);    // td0 hi | td1 lo
  g1.w[3] = (tile_d0_u8 & 0xFFFFu) << 16;                    // td1 hi=0|tile_dim0
  g1.w[4] = rows & 0xFFFFu;                                  // tile_dim1
  g1.w[5] = stride_u8;                                       // dim0_stride lo32
  i32x4 z4 = {};
#if __has_include(<hip/amd_detail/amd_gfx1250_TDM.h>)
  i32x8 z8 = {};
  __builtin_amdgcn_tensor_load_to_lds(g0.v, g1.v, z4, z4, z8, 0);
#else
  __builtin_amdgcn_tensor_load_to_lds(g0.v, g1.v, z4, z4, 0);
#endif
}

// ---------------------------------------------------------------------------
// Elementwise f32 -> bf16 cast (grid-stride).
// ---------------------------------------------------------------------------
__global__ void k_cast_bf16(const float* __restrict__ src,
                            unsigned short* __restrict__ dst, int n) {
  for (int i = blockIdx.x * blockDim.x + threadIdx.x; i < n;
       i += gridDim.x * blockDim.x)
    dst[i] = f32_to_bf16(src[i]);
}

// ---------------------------------------------------------------------------
// Small naive f32 GEMM for weight folding: C[M,N] = A[M,K] @ B[K,N].
// ---------------------------------------------------------------------------
__global__ void k_gemm_f32(const float* __restrict__ A,
                           const float* __restrict__ B,
                           float* __restrict__ C, int M, int N, int K) {
  int idx = blockIdx.x * blockDim.x + threadIdx.x;
  if (idx >= M * N) return;
  int m = idx / N, n = idx % N;
  float acc = 0.f;
  for (int k = 0; k < K; ++k) acc += A[(size_t)m * K + k] * B[(size_t)k * N + n];
  C[idx] = acc;
}

// ---------------------------------------------------------------------------
// Pack a row-major f32 [K][N] weight into the WMMA B-operand lane layout
// (bf16). 1KB block per (kt, nt) 32x16 tile; lane l owns 16 contiguous bf16:
// column n = nt*16 + (l&15), k = kt*32 + (l>=16 ? 8 : 0) + (e&7) + (e>=8 ? 16 : 0).
// ---------------------------------------------------------------------------
__global__ void k_pack_b(const float* __restrict__ src,
                         unsigned short* __restrict__ dst, int K, int N) {
  int total = K * N;
  for (int idx = blockIdx.x * blockDim.x + threadIdx.x; idx < total;
       idx += gridDim.x * blockDim.x) {
    int blk = idx >> 9;
    int within = idx & 511;
    int lane = within >> 4;
    int e = within & 15;
    int nTiles = N >> 4;
    int kt = blk / nTiles;
    int nt = blk - kt * nTiles;
    int n = nt * 16 + (lane & 15);
    int k = kt * 32 + ((lane >= 16) ? 8 : 0) + (e & 7) + ((e >= 8) ? 16 : 0);
    dst[idx] = f32_to_bf16(src[(size_t)k * N + n]);
  }
}

// ---------------------------------------------------------------------------
// A-operand fragment from row-major bf16: two aligned 16B chunks per k-tile
// (lanes<16: K 0..7,16..23; lanes>=16: 8..15,24..31).
// ---------------------------------------------------------------------------
__device__ __forceinline__ v16bf load_a_frag(const unsigned short* base) {
  v8bf lo = *(const v8bf*)(const void*)(base);
  v8bf hi = *(const v8bf*)(const void*)(base + 16);
  return __builtin_shufflevector(lo, hi, 0, 1, 2, 3, 4, 5, 6, 7,
                                 8, 9, 10, 11, 12, 13, 14, 15);
}

// ---------------------------------------------------------------------------
// WMMA bf16 GEMM with TDM-staged B: block = 128 rows x 64 cols. The 64-col
// packed-B strip is DMA'd Global->LDS in 32KB double-buffered chunks of 8
// k-tiles (tensor_load_to_lds + s_wait_tensorcnt), all 8 waves consume it via
// ds_load_b128 at immediate offsets. A stays row-major bf16 in global.
// ---------------------------------------------------------------------------
template <int K, int N>
__global__ void __launch_bounds__(256, 2)
k_wmma_tdm_gemm(const unsigned short* __restrict__ Abf,
                const unsigned short* __restrict__ Bpk,
                const float* __restrict__ bias,
                float* __restrict__ Cout, int M) {
  constexpr int KT = K / 32;                 // k-tiles
  constexpr int NT = N / 16;                 // n-tiles total
  constexpr int NG = N / 64;                 // n-strips (4 tiles each)
  constexpr int CH = 8;                      // k-tiles per LDS chunk
  constexpr int NCH = KT / CH;
  constexpr unsigned CHUNK_BYTES = CH * 4096u;  // 32 KB
  static_assert(KT % CH == 0, "K must be multiple of 256");
  __shared__ unsigned short s_b[2 * CH * 2048];  // 64 KB double buffer

  const int lane = threadIdx.x & 31;
  const int wv = threadIdx.x >> 5;
  const int bm = blockIdx.x / NG;
  const int tg = blockIdx.x - bm * NG;
  const int aoff = (lane < 16) ? 0 : 8;
  const int row = bm * 128 + wv * 16 + (lane & 15);
  const unsigned short* ap = Abf + (size_t)row * K + aoff;
  const unsigned char* gB = (const unsigned char*)Bpk + (size_t)tg * 4096u;
  const unsigned lds_base = (unsigned)(size_t)(&s_b[0]);
  constexpr unsigned STRIDE_U8 = (unsigned)NT * 128u;  // NT KB / 8

  v8f acc[4] = {};
  if (wv == 0) tdm_load_2d(lds_base, gB, 512u, CH, STRIDE_U8);

  for (int c = 0; c < NCH; ++c) {
    if (wv == 0) __builtin_amdgcn_s_wait_tensorcnt(0);
    __syncthreads();                                   // chunk c resident
    if (wv == 0 && c + 1 < NCH)
      tdm_load_2d(lds_base + (unsigned)((c + 1) & 1) * CHUNK_BYTES,
                  gB + (size_t)(c + 1) * CH * (NT * 1024u), 512u, CH,
                  STRIDE_U8);
    const unsigned short* sb = s_b + (c & 1) * (CH * 2048) + lane * 16;
    const unsigned short* ac = ap + c * CH * 32;
#pragma unroll
    for (int k8 = 0; k8 < CH; ++k8) {
      v16bf a = load_a_frag(ac + k8 * 32);
#pragma unroll
      for (int j = 0; j < 4; ++j) {
        v16bf b = *(const v16bf*)(const void*)(sb + k8 * 2048 + j * 512);
        acc[j] = __builtin_amdgcn_wmma_f32_16x16x32_bf16(
            false, a, false, b, (short)0, acc[j], false, false);
      }
    }
    __syncthreads();                                   // chunk c consumed
  }

#pragma unroll
  for (int j = 0; j < 4; ++j) {
    int n = tg * 64 + j * 16 + (lane & 15);
    float bv = bias[n];
#pragma unroll
    for (int v = 0; v < 8; ++v) {
      int m = bm * 128 + wv * 16 + v + aoff;
      Cout[(size_t)m * N + n] = acc[j][v] + bv;
    }
  }
}

// ---------------------------------------------------------------------------
// Sequential scan: one persistent workgroup (16 waves). State tile [16x512]
// (8 live batch rows) in LDS: bf16 copy = next step's WMMA A operand, f32
// copy = LN staging. The 512KB packed A-matrix streams from L2 every step; a
// memory clobber at the top of the step prevents LLVM from hoisting the
// loop-invariant B fragments out of the t-loop (which spilled to scratch).
// ---------------------------------------------------------------------------
__global__ void __launch_bounds__(512, 1)
k_scan(const float* __restrict__ u,            // [16384][512], row = b*T + t
       const unsigned short* __restrict__ Apk, // packed B-layout of A [512x512]
       const float* __restrict__ gamma, const float* __restrict__ beta,
       unsigned short* __restrict__ states) {  // bf16 [16384][512]
  __shared__ unsigned short s_bf[16 * S_DIM];  // 16 KB
  __shared__ float s_f[16 * S_DIM];            // 32 KB
  const int lane = threadIdx.x & 31;
  const int wv = threadIdx.x >> 5;             // 0..15
  const int aoff = (lane < 16) ? 0 : 8;

  for (int i = threadIdx.x; i < 16 * S_DIM; i += blockDim.x) s_bf[i] = 0;
  __syncthreads();

  float g[16], bt[16];
#pragma unroll
  for (int i = 0; i < 16; ++i) {
    g[i] = gamma[lane * 16 + i];
    bt[i] = beta[lane * 16 + i];
  }

  // Fixed per-lane pointers: B blocks at immediate offsets kt*32KB + j*1KB.
  const unsigned char* apk =
      (const unsigned char*)Apk + (size_t)(wv * 2) * 1024u + (size_t)lane * 32u;
  const unsigned short* arow = s_bf + (lane & 15) * S_DIM + aoff;
  const int n0 = (wv * 2) * 16 + (lane & 15);

  for (int t = 0; t < T_DIM; ++t) {
    // Keep per-step loads inside the loop (defeat LICM -> scratch spills).
    asm volatile("" ::: "memory");

    // ---- hoisted u_t loads (lanes<16 own live rows 0..7) ----
    float uv0[8], uv1[8];
    if (lane < 16) {
#pragma unroll
      for (int v = 0; v < 8; ++v) {
        uv0[v] = u[((size_t)v * T_DIM + t) * S_DIM + n0];
        uv1[v] = u[((size_t)v * T_DIM + t) * S_DIM + n0 + 16];
      }
    } else {
#pragma unroll
      for (int v = 0; v < 8; ++v) { uv0[v] = 0.f; uv1[v] = 0.f; }
    }

    // ---- s @ A : wave wv computes N-tiles (2*wv, 2*wv+1) ----
    v8f acc[2] = {};
#pragma unroll
    for (int kt = 0; kt < 16; ++kt) {
      v16bf a = load_a_frag(arow + kt * 32);
#pragma unroll
      for (int j = 0; j < 2; ++j) {
        v16bf b = *(const v16bf*)(const void*)(apk + kt * 32768u + j * 1024u);
        acc[j] = __builtin_amdgcn_wmma_f32_16x16x32_bf16(
            false, a, false, b, (short)0, acc[j], false, false);
      }
    }

    // ---- + u_t, stage f32 state ----
#pragma unroll
    for (int j = 0; j < 2; ++j) {
#pragma unroll
      for (int v = 0; v < 8; ++v) {
        int r = v + aoff;
        float add = (j == 0) ? uv0[v] : uv1[v];
        s_f[r * S_DIM + n0 + j * 16] = acc[j][v] + add;
      }
    }
    __syncthreads();

    // ---- LayerNorm: wave wv normalizes batch row wv (wv < 8) ----
    if (wv < B_DIM) {
      const float* rp = s_f + wv * S_DIM + lane * 16;
      float x[16], sum = 0.f, sq = 0.f;
#pragma unroll
      for (int i = 0; i < 16; ++i) {
        x[i] = rp[i];
        sum += x[i];
        sq += x[i] * x[i];
      }
#pragma unroll
      for (int o = 16; o > 0; o >>= 1) {
        sum += __shfl_xor(sum, o, 32);
        sq += __shfl_xor(sq, o, 32);
      }
      float mean = sum * (1.f / 512.f);
      float var = sq * (1.f / 512.f) - mean * mean;
      float rstd = rsqrtf(var + 1e-5f);
      unsigned short* sb = s_bf + wv * S_DIM + lane * 16;
      unsigned short* so = states + ((size_t)wv * T_DIM + t) * S_DIM + lane * 16;
#pragma unroll
      for (int i = 0; i < 16; ++i) {
        unsigned short h = f32_to_bf16((x[i] - mean) * rstd * g[i] + bt[i]);
        sb[i] = h;
        so[i] = h;
      }
    } else if (t + 1 < T_DIM) {
      int r = wv - 8;   // waves 8..15 prefetch next u rows (global_prefetch_b8)
      __builtin_prefetch(u + ((size_t)r * T_DIM + (t + 1)) * S_DIM + lane * 16,
                         0, 1);
    }
    __syncthreads();
  }
}

// ---------------------------------------------------------------------------
// Launch: fold weights -> pack -> WMMA+TDM GEMM (u) -> scan -> WMMA+TDM GEMM.
// ---------------------------------------------------------------------------
extern "C" void kernel_launch(void* const* d_in, const int* in_sizes, int n_in,
                              void* d_out, int out_size, void* d_ws,
                              size_t ws_size, hipStream_t stream) {
  const float* X      = (const float*)d_in[0];
  const float* W_e2s  = (const float*)d_in[1];
  const float* b_e2s  = (const float*)d_in[2];
  const float* A      = (const float*)d_in[3];
  const float* Bm     = (const float*)d_in[4];
  const float* Cm     = (const float*)d_in[5];
  const float* gamma  = (const float*)d_in[6];
  const float* beta   = (const float*)d_in[7];
  const float* W_s2o  = (const float*)d_in[8];
  const float* b_s2o  = (const float*)d_in[9];
  float* out = (float*)d_out;

  unsigned char* w = (unsigned char*)d_ws;
  size_t off = 0;
  unsigned short* Xbf = (unsigned short*)(w + off); off += (size_t)M_TOT * E_DIM * 2;
  float* u_buf        = (float*)(w + off);          off += (size_t)M_TOT * S_DIM * 4;
  unsigned short* stb = (unsigned short*)(w + off); off += (size_t)M_TOT * S_DIM * 2;
  float* Wf32         = (float*)(w + off);          off += (size_t)E_DIM * S_DIM * 4;
  unsigned short* WfP = (unsigned short*)(w + off); off += (size_t)E_DIM * S_DIM * 2;
  float* CW32         = (float*)(w + off);          off += (size_t)S_DIM * E_DIM * 4;
  unsigned short* CWP = (unsigned short*)(w + off); off += (size_t)S_DIM * E_DIM * 2;
  unsigned short* APk = (unsigned short*)(w + off); off += (size_t)S_DIM * S_DIM * 2;
  float* bu           = (float*)(w + off);          off += (size_t)S_DIM * 4;
  (void)off; (void)ws_size; (void)in_sizes; (void)n_in; (void)out_size;

  k_cast_bf16<<<8192, 256, 0, stream>>>(X, Xbf, M_TOT * E_DIM);

  k_gemm_f32<<<(E_DIM * S_DIM + 255) / 256, 256, 0, stream>>>(
      W_e2s, Bm, Wf32, E_DIM, S_DIM, S_DIM);
  k_gemm_f32<<<(S_DIM * E_DIM + 255) / 256, 256, 0, stream>>>(
      Cm, W_s2o, CW32, S_DIM, E_DIM, S_DIM);
  k_gemm_f32<<<(S_DIM + 255) / 256, 256, 0, stream>>>(b_e2s, Bm, bu, 1, S_DIM,
                                                      S_DIM);

  k_pack_b<<<2048, 256, 0, stream>>>(Wf32, WfP, E_DIM, S_DIM);
  k_pack_b<<<2048, 256, 0, stream>>>(CW32, CWP, S_DIM, E_DIM);
  k_pack_b<<<1024, 256, 0, stream>>>(A, APk, S_DIM, S_DIM);

  // u = Xbf @ Wf + bu : M=16384, N=512, K=1024 -> (128 m-strips)*(8 n-strips)
  k_wmma_tdm_gemm<E_DIM, S_DIM><<<128 * 8, 256, 0, stream>>>(Xbf, WfP, bu,
                                                             u_buf, M_TOT);

  k_scan<<<1, 512, 0, stream>>>(u_buf, APk, gamma, beta, stb);

  // out = states @ CW + b_s2o : M=16384, N=1024, K=512 -> 128*16 blocks
  k_wmma_tdm_gemm<S_DIM, E_DIM><<<128 * 16, 256, 0, stream>>>(stb, CWP, b_s2o,
                                                              out, M_TOT);
}
